// BahdanauAttention_41059887350367
// MI455X (gfx1250) — matrix operand order repaired and checked by reading, not compiled, not run on hardware
//
#include <hip/hip_runtime.h>
#include <math.h>
#include <stdint.h>

// ---------------------------------------------------------------------------
// Bahdanau attention, MI455X (gfx1250, wave32, WMMA).
//   B=32, S=4096, H=512
//   scores = Va . tanh(q@Wa + k@Ua); weights = softmax(scores); ctx = w@keys
// Dominant op: keys@Ua GEMM (M=131072,N=512,K=512, 68.7 GFLOP) via
// v_wmma_f32_16x16x32_bf16; f32->bf16 conversion during LDS staging.
// Fragments are loaded as 16B-aligned ds_load_b128 runs; all four B
// fragments are materialized before the WMMA burst so the four WMMAs issue
// back-to-back behind a single s_wait_dscnt.
// ---------------------------------------------------------------------------

typedef __attribute__((ext_vector_type(16))) __bf16        v16bf;
typedef __attribute__((ext_vector_type(8)))  float         v8f;
typedef __attribute__((ext_vector_type(4)))  unsigned int  v4u;

#define B_   32
#define S_   4096
#define H_   512
#define SCH  128            // S-rows per workgroup
#define NT   64             // N-tile width (columns of Ua per stage)
#define KROW 520            // padded halves per keys-LDS row (512+8) -> 1040 B (16B mult)
#define UROW 528            // padded halves per UaT-LDS row  (512+16) -> 1056 B (16B mult)
#define LDS_BYTES (SCH * KROW * 2 + NT * UROW * 2)   // 133120 + 67584 = 200704

union ABfrag { v4u q[2]; v16bf v; };

__device__ __forceinline__ uint32_t pk_bf16(float a, float b) {
  uint32_t ua = __builtin_bit_cast(uint32_t, a);
  uint32_t ub = __builtin_bit_cast(uint32_t, b);
  ua += 0x7FFFu + ((ua >> 16) & 1u);          // round-to-nearest-even
  ub += 0x7FFFu + ((ub >> 16) & 1u);
  return (ua >> 16) | (ub & 0xFFFF0000u);
}
__device__ __forceinline__ uint16_t bf16_1(float a) {
  uint32_t u = __builtin_bit_cast(uint32_t, a);
  u += 0x7FFFu + ((u >> 16) & 1u);
  return (uint16_t)(u >> 16);
}

// --------------------------- q_proj = query @ Wa ---------------------------
__global__ void qproj_kernel(const float* __restrict__ query,
                             const float* __restrict__ Wa,
                             float* __restrict__ qp) {
  const int b = blockIdx.x, tid = threadIdx.x;         // 256 threads
  float a0 = 0.f, a1 = 0.f;
  const float* q = query + (size_t)b * H_;
  for (int h = 0; h < H_; ++h) {
    const float qv = q[h];
    a0 = fmaf(qv, Wa[(size_t)h * H_ + tid],        a0);
    a1 = fmaf(qv, Wa[(size_t)h * H_ + tid + 256],  a1);
  }
  qp[(size_t)b * H_ + tid]       = a0;
  qp[(size_t)b * H_ + tid + 256] = a1;
}

// ---------- scores[b,s] = Va . tanh(qp[b] + keys[b,s]@Ua)  (WMMA) ----------
__global__ void __launch_bounds__(256)
scores_kernel(const float* __restrict__ keys, const float* __restrict__ Ua,
              const float* __restrict__ Va,  const float* __restrict__ qp,
              float* __restrict__ scores) {
  extern __shared__ unsigned char smem[];
  uint16_t* keysL = (uint16_t*)smem;                       // [SCH][KROW] bf16
  uint16_t* uaL   = (uint16_t*)(smem + SCH * KROW * 2);    // [NT][UROW]  bf16 (Ua^T)

  const int b    = blockIdx.x;
  const int s0   = blockIdx.y * SCH;
  const int tid  = threadIdx.x;          // 0..255 (8 wave32)
  const int wave = tid >> 5;
  const int lane = tid & 31;
  const int r16  = lane & 15;
  const int hi   = lane >> 4;

  // ---- stage keys[b, s0:s0+SCH, :] -> LDS as bf16 (float4 loads) ----
  {
    const float4* kg = (const float4*)(keys + ((size_t)b * S_ + s0) * H_);
    uint32_t* kL32 = (uint32_t*)keysL;
    for (int i = tid; i < SCH * (H_ / 4); i += 256) {
      const int row = i >> 7;            // 128 float4 per row
      const int c4  = i & 127;
      const float4 v = kg[(size_t)row * (H_ / 4) + c4];
      const int u32i = ((row * KROW) >> 1) + c4 * 2;
      kL32[u32i]     = pk_bf16(v.x, v.y);
      kL32[u32i + 1] = pk_bf16(v.z, v.w);
    }
  }
  __syncthreads();

  float rowAcc[8];
#pragma unroll
  for (int r = 0; r < 8; ++r) rowAcc[r] = 0.f;

  // Per-lane base pointers for fragment loads (16B-aligned runs).
  // A layout (16-bit A 16x32, ISA 7.12.2): halves {k0+hi*8 .. +7} then
  // {k0+16+hi*8 .. +7}.  B layout: same, halves {kb .. kb+7} then {kb+16 .. kb+23}, kb = k0 + 8*hi.
  const uint16_t* aRowPtr = keysL + (wave * 16 + r16) * KROW + (hi << 3);

  for (int d0 = 0; d0 < H_; d0 += NT) {
    // ---- stage Ua[:, d0:d0+NT] transposed -> uaL[n][k] ----
    for (int i = tid; i < H_ * (NT / 4); i += 256) {   // 8192 float4
      const int k  = i >> 4;             // 16 float4 per k-row
      const int n4 = i & 15;
      const float4 v = *(const float4*)(Ua + (size_t)k * H_ + d0 + n4 * 4);
      uaL[(n4 * 4 + 0) * UROW + k] = bf16_1(v.x);
      uaL[(n4 * 4 + 1) * UROW + k] = bf16_1(v.y);
      uaL[(n4 * 4 + 2) * UROW + k] = bf16_1(v.z);
      uaL[(n4 * 4 + 3) * UROW + k] = bf16_1(v.w);
    }
    __syncthreads();

    // ---- kp tile: rows [s0+wave*16,+16) x cols [d0,d0+NT), K = 512 ----
    v8f acc[4];
#pragma unroll
    for (int nt = 0; nt < 4; ++nt) acc[nt] = (v8f){0,0,0,0,0,0,0,0};

#pragma unroll 2
    for (int k0 = 0; k0 < H_; k0 += 32) {
      // A fragment: two b128 runs
      ABfrag Af;
      Af.q[0] = *(const v4u*)(aRowPtr + k0);
      Af.q[1] = *(const v4u*)(aRowPtr + k0 + 16);

      // All four B fragments up front (distinct registers -> one DS wait,
      // four back-to-back WMMAs, no WAR hazard nops)
      const int kb = k0 + (hi << 3);
      ABfrag Bf[4];
#pragma unroll
      for (int nt = 0; nt < 4; ++nt) {
        const uint16_t* bp = uaL + (nt * 16 + r16) * UROW + kb;
        Bf[nt].q[0] = *(const v4u*)(bp);
        Bf[nt].q[1] = *(const v4u*)(bp + 16);
      }
#pragma unroll
      for (int nt = 0; nt < 4; ++nt)
        acc[nt] = __builtin_amdgcn_wmma_f32_16x16x32_bf16(
            false, Af.v, false, Bf[nt].v, (short)0, acc[nt], false, false);
    }

    // ---- epilogue: rowAcc[m] += Va[d] * tanh(qp[d] + kp[m][d]) ----
#pragma unroll
    for (int nt = 0; nt < 4; ++nt) {
      const int d  = d0 + nt * 16 + r16;
      const float va = Va[d];
      const float qv = qp[(size_t)b * H_ + d];
#pragma unroll
      for (int r = 0; r < 8; ++r)
        rowAcc[r] = fmaf(va, tanhf(qv + acc[nt][r]), rowAcc[r]);
    }
    __syncthreads();   // before overwriting uaL next N-tile
  }

  // ---- reduce over n (16-lane halves); D layout: lane hi -> M = r + 8*hi ----
#pragma unroll
  for (int m = 1; m < 16; m <<= 1)
#pragma unroll
    for (int r = 0; r < 8; ++r)
      rowAcc[r] += __shfl_xor(rowAcc[r], m);

  if (r16 == 0) {
    float* sp = scores + (size_t)b * S_ + s0 + wave * 16 + hi * 8;
#pragma unroll
    for (int r = 0; r < 8; ++r) sp[r] = rowAcc[r];
  }
}

// ------------- softmax per batch; write weights; zero ctx region -----------
__global__ void softmax_kernel(const float* __restrict__ scores,
                               float* __restrict__ out) {
  __shared__ float red[256];
  const int b = blockIdx.x, tid = threadIdx.x;
  const float* sc = scores + (size_t)b * S_;

  float m = -3.4e38f;
  for (int i = tid; i < S_; i += 256) m = fmaxf(m, sc[i]);
  red[tid] = m; __syncthreads();
  for (int st = 128; st > 0; st >>= 1) {
    if (tid < st) red[tid] = fmaxf(red[tid], red[tid + st]);
    __syncthreads();
  }
  const float gmax = red[0]; __syncthreads();

  float s = 0.f;
  for (int i = tid; i < S_; i += 256) s += __expf(sc[i] - gmax);
  red[tid] = s; __syncthreads();
  for (int st = 128; st > 0; st >>= 1) {
    if (tid < st) red[tid] += red[tid + st];
    __syncthreads();
  }
  const float inv = 1.0f / red[0];

  float* w = out + (size_t)B_ * H_ + (size_t)b * S_;
  for (int i = tid; i < S_; i += 256) w[i] = __expf(sc[i] - gmax) * inv;
  for (int i = tid; i < H_; i += 256) out[(size_t)b * H_ + i] = 0.f;  // ctx=0
}

// ----------------- context[b] = sum_s w[b,s] * keys[b,s,:] -----------------
__global__ void context_kernel(const float* __restrict__ keys,
                               const float* __restrict__ outw,
                               float* __restrict__ out) {
  const int b = blockIdx.x, ch = blockIdx.y, tid = threadIdx.x;   // 128 thr
  const float*  w  = outw + (size_t)B_ * H_ + (size_t)b * S_ + (size_t)ch * 512;
  const float4* kg = (const float4*)(keys + ((size_t)b * S_ + (size_t)ch * 512) * H_);
  float4 a = {0.f, 0.f, 0.f, 0.f};
  for (int s = 0; s < 512; ++s) {
    const float  ws = w[s];
    const float4 kv = kg[(size_t)s * (H_ / 4) + tid];
    a.x = fmaf(ws, kv.x, a.x); a.y = fmaf(ws, kv.y, a.y);
    a.z = fmaf(ws, kv.z, a.z); a.w = fmaf(ws, kv.w, a.w);
  }
  float* ctx = out + (size_t)b * H_ + tid * 4;
  atomicAdd(ctx + 0, a.x); atomicAdd(ctx + 1, a.y);
  atomicAdd(ctx + 2, a.z); atomicAdd(ctx + 3, a.w);
}

// ---------------------------------------------------------------------------
extern "C" void kernel_launch(void* const* d_in, const int* in_sizes, int n_in,
                              void* d_out, int out_size, void* d_ws, size_t ws_size,
                              hipStream_t stream) {
  const float* query = (const float*)d_in[0];   // [32,1,512]
  const float* keys  = (const float*)d_in[1];   // [32,4096,512]
  const float* Wa    = (const float*)d_in[2];   // [512,512]
  const float* Ua    = (const float*)d_in[3];   // [512,512]
  const float* Va    = (const float*)d_in[4];   // [512,1]
  float* out = (float*)d_out;                   // ctx[32*512] ++ weights[32*4096]

  float* qp     = (float*)d_ws;                 // [32,512]
  float* scores = qp + (size_t)B_ * H_;         // [32,4096]

  qproj_kernel<<<B_, 256, 0, stream>>>(query, Wa, qp);
  scores_kernel<<<dim3(B_, S_ / SCH), 256, LDS_BYTES, stream>>>(keys, Ua, Va, qp, scores);
  softmax_kernel<<<B_, 256, 0, stream>>>(scores, out);
  context_kernel<<<dim3(B_, 8), 128, 0, stream>>>(keys, out, out);
}
